// InvariantPoseAttention_7988639171222
// MI455X (gfx1250) — matrix operand order repaired and checked by reading, not compiled, not run on hardware
//
#include <hip/hip_runtime.h>
#include <cstdint>

typedef __attribute__((ext_vector_type(16))) _Float16 v16h;
typedef __attribute__((ext_vector_type(8)))  float    v8f;

#define BB 8
#define NN 1024
#define DIMD 512
#define HH 8
#define DHD 64
#define INNER_ 512

__device__ __forceinline__ v8f wmma16(v16h a, v16h b, v8f c) {
  // D = A(16x32,f16) * B(32x16,f16) + C(16x16,f32)
  return __builtin_amdgcn_wmma_f32_16x16x32_f16(false, a, false, b, (short)0, c,
                                                false, false);
}

// A-fragment (16x32, row-major M x K source, stride lda halves).
// Lane m=lane&15 holds row m; halves h = K(g*16+h).
// One contiguous 32B span per lane -> ds_load_b128 / global_load_b128 pairs.
__device__ __forceinline__ v16h load_frag_a(const _Float16* src, int lda, int lane) {
  const int m = lane & 15;
  const int g = (lane >> 4) & 1;
  const _Float16* p = src + m * lda + g * 16;
  v16h a;
#pragma unroll
  for (int h = 0; h < 16; ++h) a[h] = p[h];
  return a;
}

// B-fragment from an N-major source (Bt[n][k], stride ldk halves).
// Lane n=lane&15 holds column n; halves h = K(h + 16*kg).  The 16 halves are
// one contiguous 32B span per lane -> two ds_load_b128.
__device__ __forceinline__ v16h load_frag_bt(const _Float16* src, int ldk, int lane) {
  const int n = lane & 15;
  const int kg = (lane >> 4) & 1;
  const _Float16* p = src + n * ldk + kg * 16;
  v16h b;
#pragma unroll
  for (int h = 0; h < 16; ++h) b[h] = p[h];
  return b;
}

// ---------------------------------------------------------------- conversions
__global__ __launch_bounds__(256) void cvt_f32_f16(const float* __restrict__ src,
                                                   _Float16* __restrict__ dst, int n) {
  int i = blockIdx.x * 256 + threadIdx.x;
  if (i < n) dst[i] = (_Float16)src[i];
}

// dst[n][k] = (f16) src[k][n]   (weights: transpose once so GEMM B tiles are
// N-major and B-fragments read contiguously)
__global__ __launch_bounds__(256) void cvt_transpose_f16(
    const float* __restrict__ src, _Float16* __restrict__ dst, int rows, int cols) {
  int i = blockIdx.x * 256 + threadIdx.x;
  if (i < rows * cols) {
    int n = i / rows, k = i - n * rows;
    dst[i] = (_Float16)src[(size_t)k * cols + n];
  }
}

// ------------------------------------------------------------ generic WMMA GEMM
// C[M,N] f32 = A[M,K] f16 * Bt[N,K] f16 + bias[N].  Block tile 128x128, 8 waves,
// wave tile 32x64, K step 32, double-buffered LDS.
__global__ __launch_bounds__(256) void gemm_f16_wmma(
    const _Float16* __restrict__ A, const _Float16* __restrict__ Bt,
    const float* __restrict__ bias, float* __restrict__ C, int M, int N, int K) {
  __shared__ _Float16 As[2][128][40];   // 80B rows: 16B aligned, conflict-free
  __shared__ _Float16 Bs[2][128][40];   // N-major B tile
  const int tid = threadIdx.x;
  const int lane = tid & 31;
  const int wave = tid >> 5;
  const int wm = wave & 3;              // 4 waves along M
  const int wn = wave >> 2;             // 2 waves along N
  const int mblk = blockIdx.y * 128;
  const int nblk = blockIdx.x * 128;

  const int sr = tid >> 1, sc = (tid & 1) * 16;   // staging coords (16 halves)
  auto stage = [&](int k0, int buf) {
    const uint4* ga = (const uint4*)(A + (size_t)(mblk + sr) * K + k0 + sc);
    uint4* sa = (uint4*)&As[buf][sr][sc];
    sa[0] = ga[0]; sa[1] = ga[1];
    const uint4* gb = (const uint4*)(Bt + (size_t)(nblk + sr) * K + k0 + sc);
    uint4* sb = (uint4*)&Bs[buf][sr][sc];
    sb[0] = gb[0]; sb[1] = gb[1];
  };

  v8f zero{};
  v8f acc[2][4];
#pragma unroll
  for (int i = 0; i < 2; ++i)
#pragma unroll
    for (int j = 0; j < 4; ++j) acc[i][j] = zero;

  stage(0, 0);
  __syncthreads();
  for (int k0 = 0; k0 < K; k0 += 32) {
    const int cur = (k0 >> 5) & 1;
    if (k0 + 32 < K) stage(k0 + 32, cur ^ 1);   // overlap staging with WMMAs
    v16h a0 = load_frag_a(&As[cur][wm * 32][0], 40, lane);
    v16h a1 = load_frag_a(&As[cur][wm * 32 + 16][0], 40, lane);
#pragma unroll
    for (int j = 0; j < 4; ++j) {
      v16h bf = load_frag_bt(&Bs[cur][wn * 64 + j * 16][0], 40, lane);
      acc[0][j] = wmma16(a0, bf, acc[0][j]);
      acc[1][j] = wmma16(a1, bf, acc[1][j]);
    }
    __syncthreads();
  }

  const int ncol = lane & 15;
  const int mbase = (lane >> 4) * 8;
#pragma unroll
  for (int i = 0; i < 2; ++i)
#pragma unroll
    for (int j = 0; j < 4; ++j) {
      int n = nblk + wn * 64 + j * 16 + ncol;
      float bv = bias[n];
#pragma unroll
      for (int r = 0; r < 8; ++r) {
        int m = mblk + wm * 32 + i * 16 + mbase + r;
        C[(size_t)m * N + n] = acc[i][j][r] + bv;
      }
    }
}

// ------------------------------------------------- SE(3) pose on q/k/v groups
// One thread per (b,n,h,group-of-4).  q <- scale * P_qinv^T q ; k,v <- P_kv {k,v}.
// q,k written [B,H,N,DH]; v written TRANSPOSED [B,H,DH,N] so the attention
// P*V B-fragments read contiguous key spans.
__global__ __launch_bounds__(256) void pose_qkv_kernel(
    const float* __restrict__ qkv, const float* __restrict__ rot,
    const float* __restrict__ trans, _Float16* __restrict__ qh,
    _Float16* __restrict__ kh, _Float16* __restrict__ vt) {
  int idx = blockIdx.x * 256 + threadIdx.x;
  const int g = idx & 15;
  const int h = (idx >> 4) & 7;
  const int n = (idx >> 7) & 1023;
  const int b = idx >> 17;
  const int tok = b * NN + n;
  const float* Rp = rot + (size_t)tok * 9;
  const float* tp = trans + (size_t)tok * 3;
  float R[3][3], tt[3], ti[3];
#pragma unroll
  for (int i = 0; i < 3; ++i) {
    tt[i] = tp[i];
#pragma unroll
    for (int j = 0; j < 3; ++j) R[i][j] = Rp[i * 3 + j];
  }
#pragma unroll
  for (int i = 0; i < 3; ++i)
    ti[i] = -(R[0][i] * tt[0] + R[1][i] * tt[1] + R[2][i] * tt[2]);  // -R^T t

  const size_t tb = (size_t)tok * (3 * INNER_);
  const int d0 = h * DHD + g * 4;
  float q[4], k[4], v[4];
#pragma unroll
  for (int j = 0; j < 4; ++j) {
    q[j] = qkv[tb + d0 + j];
    k[j] = qkv[tb + INNER_ + d0 + j];
    v[j] = qkv[tb + 2 * INNER_ + d0 + j];
  }
  float qo[4], ko[4], vo[4];
#pragma unroll
  for (int i = 0; i < 3; ++i) {
    qo[i] = R[i][0] * q[0] + R[i][1] * q[1] + R[i][2] * q[2];
    ko[i] = R[i][0] * k[0] + R[i][1] * k[1] + R[i][2] * k[2] + tt[i] * k[3];
    vo[i] = R[i][0] * v[0] + R[i][1] * v[1] + R[i][2] * v[2] + tt[i] * v[3];
  }
  qo[3] = ti[0] * q[0] + ti[1] * q[1] + ti[2] * q[2] + q[3];
  ko[3] = k[3];
  vo[3] = v[3];

  const float scale = 0.125f;  // DH^-0.5 folded into q
  const size_t ob = ((size_t)(b * HH + h) * NN + n) * DHD + g * 4;
  const size_t vb = (size_t)(b * HH + h) * DHD * NN + (size_t)(g * 4) * NN + n;
#pragma unroll
  for (int j = 0; j < 4; ++j) {
    qh[ob + j] = (_Float16)(qo[j] * scale);
    kh[ob + j] = (_Float16)ko[j];
    vt[vb + (size_t)j * NN] = (_Float16)vo[j];
  }
}

// ------------------------------------------------------------- flash attention
// One block per (b,h, 64-query tile).  4 waves x 16 query rows.  K chunk staged
// row-major (its natural layout IS the N-major B layout for Q*K^T); V staged
// from the pre-transposed global tensor.  Double-buffered chunks.
__global__ __launch_bounds__(128) void ipa_attention(
    const _Float16* __restrict__ Q, const _Float16* __restrict__ Kg,
    const _Float16* __restrict__ Vtg, float* __restrict__ O) {
  __shared__ _Float16 Ks[2][64][72];     // [key][d]  (N-major for S = Q K^T)
  __shared__ _Float16 Vt[2][64][72];     // [d][key]  (N-major for O = P V)
  __shared__ _Float16 Ps[4][16][72];     // per-wave P tile (C-layout -> A-layout)
  const int tid = threadIdx.x;
  const int lane = tid & 31;
  const int wave = tid >> 5;
  const int qblk = blockIdx.x % (NN / 64);
  const int bh = blockIdx.x / (NN / 64);
  const size_t base = (size_t)bh * NN * DHD;   // q,k,o base
  const size_t vbase = (size_t)bh * DHD * NN;  // v^T base
  const int q0 = qblk * 64 + wave * 16;

  auto stage = [&](int kc, int buf) {
#pragma unroll
    for (int i = tid; i < 512; i += 128) {   // 64x64 halves as uint4 chunks
      int row = i >> 3, c = (i & 7) * 8;
      *(uint4*)&Ks[buf][row][c] =
          *(const uint4*)(Kg + base + (size_t)(kc + row) * DHD + c);
      *(uint4*)&Vt[buf][row][c] =
          *(const uint4*)(Vtg + vbase + (size_t)row * NN + kc + c);
    }
  };

  // Q tile as two A-fragments (d 0..31 / 32..63), held for whole kernel
  v16h aq0 = load_frag_a(Q + base + (size_t)q0 * DHD, DHD, lane);
  v16h aq1 = load_frag_a(Q + base + (size_t)q0 * DHD + 32, DHD, lane);

  v8f zero{};
  v8f acco[4];
#pragma unroll
  for (int t = 0; t < 4; ++t) acco[t] = zero;
  float mrow[8], lrow[8];
#pragma unroll
  for (int r = 0; r < 8; ++r) { mrow[r] = -1e30f; lrow[r] = 0.f; }

  stage(0, 0);
  __syncthreads();
  for (int kc = 0; kc < NN; kc += 64) {
    const int cur = (kc >> 6) & 1;
    if (kc + 64 < NN) stage(kc + 64, cur ^ 1);   // prefetch next chunk

    // S = Q K^T : 4 key tiles of 16, chained over d (2 x 32)
    v8f s[4];
#pragma unroll
    for (int j = 0; j < 4; ++j) {
      s[j] = zero;
      s[j] = wmma16(aq0, load_frag_bt(&Ks[cur][j * 16][0], 72, lane), s[j]);
      s[j] = wmma16(aq1, load_frag_bt(&Ks[cur][j * 16][32], 72, lane), s[j]);
    }

    // Online softmax.  Accumulator slot r <-> row (8*(lane>=16)+r): shared by
    // the 16 lanes of each half-wave -> shfl_xor reductions of width 16.
    float alpha[8];
#pragma unroll
    for (int r = 0; r < 8; ++r) {
      float mx = fmaxf(fmaxf(s[0][r], s[1][r]), fmaxf(s[2][r], s[3][r]));
#pragma unroll
      for (int off = 1; off < 16; off <<= 1) mx = fmaxf(mx, __shfl_xor(mx, off, 16));
      float mnew = fmaxf(mrow[r], mx);
      alpha[r] = __expf(mrow[r] - mnew);
      float rs = 0.f;
#pragma unroll
      for (int j = 0; j < 4; ++j) {
        float p = __expf(s[j][r] - mnew);
        s[j][r] = p;
        rs += p;
      }
#pragma unroll
      for (int off = 1; off < 16; off <<= 1) rs += __shfl_xor(rs, off, 16);
      lrow[r] = lrow[r] * alpha[r] + rs;
      mrow[r] = mnew;
    }
#pragma unroll
    for (int t = 0; t < 4; ++t)
#pragma unroll
      for (int r = 0; r < 8; ++r) acco[t][r] *= alpha[r];

    // C-layout P -> per-wave LDS -> A-layout fragments (DS ops in-order per wave)
    {
      const int pm = (lane >> 4) * 8;
      const int pc = lane & 15;
#pragma unroll
      for (int j = 0; j < 4; ++j)
#pragma unroll
        for (int r = 0; r < 8; ++r)
          Ps[wave][pm + r][j * 16 + pc] = (_Float16)s[j][r];
    }
    asm volatile("" ::: "memory");
    v16h ap0 = load_frag_a(&Ps[wave][0][0], 72, lane);
    v16h ap1 = load_frag_a(&Ps[wave][0][32], 72, lane);
#pragma unroll
    for (int t = 0; t < 4; ++t) {
      acco[t] = wmma16(ap0, load_frag_bt(&Vt[cur][t * 16][0], 72, lane), acco[t]);
      acco[t] = wmma16(ap1, load_frag_bt(&Vt[cur][t * 16][32], 72, lane), acco[t]);
    }
    __syncthreads();
  }

  const int pc = lane & 15;
  const int mb = (lane >> 4) * 8;
#pragma unroll
  for (int r = 0; r < 8; ++r) {
    float inv = 1.0f / lrow[r];
#pragma unroll
    for (int t = 0; t < 4; ++t)
      O[base + (size_t)(q0 + mb + r) * DHD + t * 16 + pc] = acco[t][r] * inv;
  }
}

// ---------------------------------------- inverse pose on output + f16 convert
__global__ __launch_bounds__(256) void pose_out_kernel(
    const float* __restrict__ o, const float* __restrict__ rot,
    const float* __restrict__ trans, _Float16* __restrict__ oh) {
  int idx = blockIdx.x * 256 + threadIdx.x;
  const int g = idx & 15;
  const int h = (idx >> 4) & 7;
  const int n = (idx >> 7) & 1023;
  const int b = idx >> 17;
  const int tok = b * NN + n;
  const float* Rp = rot + (size_t)tok * 9;
  const float* tp = trans + (size_t)tok * 3;
  float R[3][3], tt[3], ti[3];
#pragma unroll
  for (int i = 0; i < 3; ++i) {
    tt[i] = tp[i];
#pragma unroll
    for (int j = 0; j < 3; ++j) R[i][j] = Rp[i * 3 + j];
  }
#pragma unroll
  for (int i = 0; i < 3; ++i)
    ti[i] = -(R[0][i] * tt[0] + R[1][i] * tt[1] + R[2][i] * tt[2]);

  const size_t ib = ((size_t)(b * HH + h) * NN + n) * DHD + g * 4;
  float x[4];
#pragma unroll
  for (int j = 0; j < 4; ++j) x[j] = o[ib + j];
  float y[4];
#pragma unroll
  for (int i = 0; i < 3; ++i)  // R^T x + t_inv * x3
    y[i] = R[0][i] * x[0] + R[1][i] * x[1] + R[2][i] * x[2] + ti[i] * x[3];
  y[3] = x[3];

  const size_t ob = (size_t)tok * INNER_ + h * DHD + g * 4;
#pragma unroll
  for (int j = 0; j < 4; ++j) oh[ob + j] = (_Float16)y[j];
}

// ---------------------------------------------------------------------- launch
extern "C" void kernel_launch(void* const* d_in, const int* in_sizes, int n_in,
                              void* d_out, int out_size, void* d_ws, size_t ws_size,
                              hipStream_t stream) {
  (void)in_sizes; (void)n_in; (void)out_size; (void)ws_size;
  const float* x     = (const float*)d_in[0];
  const float* rot   = (const float*)d_in[1];
  const float* trans = (const float*)d_in[2];
  const float* w_qkv = (const float*)d_in[3];
  const float* b_qkv = (const float*)d_in[4];
  const float* w_out = (const float*)d_in[5];
  const float* b_out = (const float*)d_in[6];
  float* out = (float*)d_out;
  char* ws = (char*)d_ws;

  constexpr size_t NX    = (size_t)BB * NN * DIMD;      // 4,194,304
  constexpr size_t NWQKV = (size_t)DIMD * 3 * INNER_;   //   786,432
  constexpr size_t NWOUT = (size_t)INNER_ * DIMD;       //   262,144
  constexpr size_t NHEAD = (size_t)BB * HH * NN * DHD;  // 4,194,304

  constexpr size_t OFF_XH   = 0;
  constexpr size_t OFF_WQKV = OFF_XH + NX * 2;
  constexpr size_t OFF_WOUT = OFF_WQKV + NWQKV * 2;
  constexpr size_t OFF_QH   = OFF_WOUT + NWOUT * 2;
  constexpr size_t OFF_KH   = OFF_QH + NHEAD * 2;
  constexpr size_t OFF_VT   = OFF_KH + NHEAD * 2;
  constexpr size_t OFF_OH   = OFF_VT + NHEAD * 2;
  constexpr size_t OFF_QKV  = OFF_OH + (size_t)BB * NN * INNER_ * 2;
  constexpr size_t OFF_O    = OFF_QKV;  // attn out reuses dead qkv-fp32 region

  _Float16* xh    = (_Float16*)(ws + OFF_XH);
  _Float16* wqkvt = (_Float16*)(ws + OFF_WQKV);  // [3*INNER][DIM]
  _Float16* woutt = (_Float16*)(ws + OFF_WOUT);  // [DIM][INNER]
  _Float16* qh    = (_Float16*)(ws + OFF_QH);
  _Float16* kh    = (_Float16*)(ws + OFF_KH);
  _Float16* vt    = (_Float16*)(ws + OFF_VT);
  _Float16* oh    = (_Float16*)(ws + OFF_OH);
  float*    qkv   = (float*)(ws + OFF_QKV);
  float*    o     = (float*)(ws + OFF_O);

  cvt_f32_f16<<<(int)(NX / 256), 256, 0, stream>>>(x, xh, (int)NX);
  cvt_transpose_f16<<<(int)(NWQKV / 256), 256, 0, stream>>>(w_qkv, wqkvt, 512, 1536);
  cvt_transpose_f16<<<(int)(NWOUT / 256), 256, 0, stream>>>(w_out, woutt, 512, 512);

  // qkv = x @ w_qkv + b_qkv   (M=8192, N=1536, K=512)
  gemm_f16_wmma<<<dim3(1536 / 128, 8192 / 128), 256, 0, stream>>>(
      xh, wqkvt, b_qkv, qkv, 8192, 1536, 512);

  // pose rotate q/k/v ; q,k -> [B,H,N,DH], v -> [B,H,DH,N] (transposed)
  pose_qkv_kernel<<<(BB * NN * HH * 16) / 256, 256, 0, stream>>>(
      qkv, rot, trans, qh, kh, vt);

  // attention: 1 block per (b,h, 64 query rows)
  ipa_attention<<<BB * HH * (NN / 64), 128, 0, stream>>>(qh, kh, vt, o);

  // inverse pose + merge heads to [B,N,INNER] f16
  pose_out_kernel<<<(BB * NN * HH * 16) / 256, 256, 0, stream>>>(o, rot, trans, oh);

  // out = oh @ w_out + b_out   (M=8192, N=512, K=512)
  gemm_f16_wmma<<<dim3(512 / 128, 8192 / 128), 256, 0, stream>>>(
      oh, woutt, b_out, out, 8192, 512, 512);
}
